// PrunedConv_73890617361046
// MI455X (gfx1250) — compile-verified, matrix operand order checked by reading, not run on hardware
//
#include <hip/hip_runtime.h>

// Problem constants (from reference)
#define BB    16
#define HH    64
#define WW    64
#define CIN   256
#define DMUL  4
#define COUT  512
#define DWC   (CIN * DMUL)   // 1024 depthwise channels == GEMM K
#define TILEW 16             // pixels per M tile
#define MT    4              // M tiles per workgroup (64 pixels = full W row)
#define NT    4              // N tiles per wave (64 channels)
#define YSTRIDE 1032         // f16 elems per LDS row: 1024 + 8 pad -> 2064B (bank-friendly)

typedef __attribute__((ext_vector_type(16))) _Float16 v16h;
typedef __attribute__((ext_vector_type(8)))  _Float16 v8h;
typedef __attribute__((ext_vector_type(4)))  _Float16 v4h;
typedef __attribute__((ext_vector_type(8)))  float    v8f;

// ---------------------------------------------------------------------------
// Setup: build S^T (COUT x DWC, f16, 0/1) in workspace.  Rebuilt every call
// (graph-capture safe, no static state).
// ---------------------------------------------------------------------------
__global__ void zero_st_kernel(unsigned int* p, int nwords) {
    int i = blockIdx.x * blockDim.x + threadIdx.x;
    if (i < nwords) p[i] = 0u;
}

__global__ void fill_st_kernel(const int* __restrict__ sidx,
                               _Float16* __restrict__ st) {
    int j = blockIdx.x * blockDim.x + threadIdx.x;   // 0..DWC-1
    if (j < DWC) {
        int n = sidx[j];                              // 0..COUT-1
        st[(size_t)n * DWC + j] = (_Float16)1.0f;     // S^T[n][j] = 1
    }
}

// ---------------------------------------------------------------------------
// Fused depthwise-conv (VALU) -> channel scatter as f16 WMMA GEMM -> bias+ReLU
// One 256-thread block (8 waves) per (b,h) row: 64 pixels.
//   Phase 1: y[64][1024] (f16) into LDS, WMMA-A-fragment friendly layout.
//   Phase 2: wave w computes out[64 px][64 ch] as 4x4 tile grid of
//            v_wmma_f32_16x16x32_f16; each B fragment amortized over 4 M tiles.
// ---------------------------------------------------------------------------
__global__ __launch_bounds__(256)
void dwconv_scatter_wmma(const float* __restrict__ x,
                         const float* __restrict__ kern,
                         const float* __restrict__ bias,
                         const _Float16* __restrict__ st,
                         float* __restrict__ out) {
    __shared__ __align__(16) _Float16 ylds[MT * TILEW * YSTRIDE];   // 132 KB

    const int tid  = threadIdx.x;
    const int lane = tid & 31;
    const int wave = tid >> 5;                // 0..7

    const int h = blockIdx.x & (HH - 1);
    const int b = blockIdx.x >> 6;

    // ---------------- Phase 1: depthwise 3x3 conv, y -> LDS (f16) ----------
    // 256 threads == 256 input channels; each thread does all 64 pixels
    // in four 16-pixel chunks.
    {
        const int i = tid;                    // input channel
        for (int ps = 0; ps < MT; ++ps) {
            const int w0 = ps * TILEW;
            float acc[DMUL][TILEW];
            #pragma unroll
            for (int l = 0; l < DMUL; ++l)
                #pragma unroll
                for (int p = 0; p < TILEW; ++p) acc[l][p] = 0.0f;

            #pragma unroll
            for (int kh = 0; kh < 3; ++kh) {
                const int hh = h + kh - 1;
                const bool vh = (hh >= 0) && (hh < HH);
                float xr[TILEW + 2];
                #pragma unroll
                for (int t = 0; t < TILEW + 2; ++t) {
                    const int wc = w0 - 1 + t;
                    float v = 0.0f;
                    if (vh && wc >= 0 && wc < WW)
                        v = x[(((size_t)b * HH + hh) * WW + wc) * CIN + i];
                    xr[t] = v;
                }
                #pragma unroll
                for (int kw = 0; kw < 3; ++kw) {
                    // kernel[kh][kw][i][0..3] is 16B contiguous -> one b128 load
                    const float4 w4 =
                        reinterpret_cast<const float4*>(kern)[(kh * 3 + kw) * CIN + i];
                    const float wv[DMUL] = {w4.x, w4.y, w4.z, w4.w};
                    #pragma unroll
                    for (int p = 0; p < TILEW; ++p) {
                        const float xv = xr[p + kw];
                        #pragma unroll
                        for (int l = 0; l < DMUL; ++l)
                            acc[l][p] = fmaf(xv, wv[l], acc[l][p]);
                    }
                }
            }
            // Store y[w0+p][i*4 + l] as packed f16 (8B per pixel).
            #pragma unroll
            for (int p = 0; p < TILEW; ++p) {
                v4h v;
                v[0] = (_Float16)acc[0][p];
                v[1] = (_Float16)acc[1][p];
                v[2] = (_Float16)acc[2][p];
                v[3] = (_Float16)acc[3][p];
                *reinterpret_cast<v4h*>(&ylds[(w0 + p) * YSTRIDE + i * DMUL]) = v;
            }
        }
    }
    __syncthreads();

    // ---------------- Phase 2: out = y @ S  via v_wmma_f32_16x16x32_f16 ----
    const int n0   = wave * (NT * 16);        // this wave's channel base
    const int frag = lane & 15;               // A: pixel row (M); B: column (N)
    const int sel  = (lane >> 4) * 8;         // per-lane-half K-element select

    v8f accv[MT][NT];
    const v8f vzero = {};
    #pragma unroll
    for (int mt = 0; mt < MT; ++mt)
        #pragma unroll
        for (int nt = 0; nt < NT; ++nt) accv[mt][nt] = vzero;

    for (int kk = 0; kk < DWC; kk += 32) {
        // A fragments for all 4 M tiles (ISA 7.12.2 16-bit A layout:
        // lanes 0-15 hold K {kk..kk+7, kk+16..kk+23}; lanes 16-31 the rest).
        v16h a[MT];
        #pragma unroll
        for (int mt = 0; mt < MT; ++mt) {
            const _Float16* pa =
                &ylds[(mt * TILEW + frag) * YSTRIDE] + kk + sel;
            const v8h alo = *reinterpret_cast<const v8h*>(pa);
            const v8h ahi = *reinterpret_cast<const v8h*>(pa + 16);
            a[mt] = __builtin_shufflevector(alo, ahi,
                0, 1, 2, 3, 4, 5, 6, 7, 8, 9, 10, 11, 12, 13, 14, 15);
        }
        #pragma unroll
        for (int nt = 0; nt < NT; ++nt) {
            // B fragment (32x16 f16) from S^T[n][k]; reused across 4 M tiles.
            const _Float16* pb =
                st + (size_t)(n0 + nt * 16 + frag) * DWC + kk + sel;
            const v8h blo = *reinterpret_cast<const v8h*>(pb);
            const v8h bhi = *reinterpret_cast<const v8h*>(pb + 16);
            const v16h bm = __builtin_shufflevector(blo, bhi,
                0, 1, 2, 3, 4, 5, 6, 7, 8, 9, 10, 11, 12, 13, 14, 15);

            #pragma unroll
            for (int mt = 0; mt < MT; ++mt)
                accv[mt][nt] = __builtin_amdgcn_wmma_f32_16x16x32_f16(
                    false, a[mt], false, bm, (short)0, accv[mt][nt], false, false);
        }
    }

    // ---------------- Epilogue: bias + ReLU + store -------------------------
    // C/D layout: VGPR r, lanes 0-15 -> (M=r, N=lane); lanes 16-31 -> (M=8+r).
    const int mhalf = (lane >> 4) * 8;
    #pragma unroll
    for (int nt = 0; nt < NT; ++nt) {
        const int ch = n0 + nt * 16 + frag;
        const float bv = bias[ch];
        #pragma unroll
        for (int mt = 0; mt < MT; ++mt) {
            #pragma unroll
            for (int r = 0; r < 8; ++r) {
                const int p = mt * TILEW + r + mhalf;   // pixel in row
                float v = accv[mt][nt][r] + bv;
                v = v > 0.0f ? v : 0.0f;
                out[(((size_t)b * HH + h) * WW + p) * COUT + ch] = v;
            }
        }
    }
}

// ---------------------------------------------------------------------------
extern "C" void kernel_launch(void* const* d_in, const int* in_sizes, int n_in,
                              void* d_out, int out_size, void* d_ws, size_t ws_size,
                              hipStream_t stream) {
    const float* x    = (const float*)d_in[0];
    const float* kern = (const float*)d_in[1];
    const float* bias = (const float*)d_in[2];
    const int*   sidx = (const int*)d_in[3];
    float* out = (float*)d_out;

    _Float16* st = (_Float16*)d_ws;                  // S^T: COUT x DWC f16 (1 MiB)
    const int stwords = COUT * DWC / 2;              // as 32-bit words

    zero_st_kernel<<<(stwords + 255) / 256, 256, 0, stream>>>(
        (unsigned int*)d_ws, stwords);
    fill_st_kernel<<<(DWC + 255) / 256, 256, 0, stream>>>(sidx, st);

    const int nrows = BB * HH;                       // 1024 workgroups
    dwconv_scatter_wmma<<<nrows, 256, 0, stream>>>(x, kern, bias, st, out);
}
